// DeepGATNet_57767310131502
// MI455X (gfx1250) — compile-verified, matrix operand order checked by reading, not run on hardware
//
#include <hip/hip_runtime.h>
#include <hip/hip_bf16.h>
#include <float.h>

// ---------------------------------------------------------------------------
// Types for WMMA intrinsics (CDNA5 / gfx1250, wave32)
// ---------------------------------------------------------------------------
typedef __attribute__((ext_vector_type(8)))  float        v8f;
typedef __attribute__((ext_vector_type(16))) __bf16       v16bf;
typedef __attribute__((ext_vector_type(4)))  unsigned int v4u;

union FragU {                 // 32 bytes: two b128 LDS loads -> one WMMA operand
  v4u   u[2];
  v16bf b;
};
union Pack4 {                 // 4 bf16 packed for one b64 LDS store
  __bf16             h[4];
  unsigned long long u64;
};

#define TILE_M 128
#define TILE_K 32
#define LDT    40   // tile row stride in bf16 elems: 80 B, 16 B aligned, odd bank stride

// ---------------------------------------------------------------------------
// Tiled GEMM: out[Nrows,Hout] = A[Nrows,K] @ W[K,Hout]   (fp32 in/out,
// bf16 WMMA compute, f32 accumulate). Block = 256 threads = 8 waves;
// wave w owns rows [w*16, w*16+16) and NCT 16x16 accumulator tiles.
// A staged row-major, B staged column-major, so every WMMA fragment is
// exactly two contiguous ds_load_b128 per the ISA 16-bit operand layout:
// lane (M, hs) needs K in [8hs,8hs+8) U [16+8hs,16+8hs+8).
// ---------------------------------------------------------------------------
template<int NCT>
__global__ __launch_bounds__(256)
void k_gemm_wmma(const float* __restrict__ A, const float* __restrict__ W,
                 float* __restrict__ out, int Nrows, int K, int Hout)
{
  __shared__ __bf16 As[TILE_M * LDT];      // row-major:    As[row*LDT + k]
  __shared__ __bf16 Bs[16 * NCT * LDT];    // column-major: Bs[col*LDT + k]

  const int tid     = threadIdx.x;
  const int lane    = tid & 31;
  const int wave    = tid >> 5;
  const int rowBase = blockIdx.x * TILE_M;
  const int m0      = wave * 16;
  const int M       = lane & 15;
  const int hs      = lane >> 4;

  v8f acc[NCT];
  #pragma unroll
  for (int n = 0; n < NCT; ++n) acc[n] = (v8f){0.f,0.f,0.f,0.f,0.f,0.f,0.f,0.f};

  for (int k0 = 0; k0 < K; k0 += TILE_K) {
    // --- stage A tile (128x32): float4 global loads, packed b64 LDS stores
    #pragma unroll
    for (int it = 0; it < 4; ++it) {
      int i   = (tid + it * 256) * 4;          // element index in 128x32 tile
      int r   = i >> 5, c = i & 31;
      int row = rowBase + r;
      float4 v = make_float4(0.f, 0.f, 0.f, 0.f);
      if (row < Nrows)
        v = *(const float4*)(A + (size_t)row * K + (k0 + c));
      Pack4 p;
      p.h[0] = (__bf16)v.x; p.h[1] = (__bf16)v.y;
      p.h[2] = (__bf16)v.z; p.h[3] = (__bf16)v.w;
      *(unsigned long long*)&As[r * LDT + c] = p.u64;
    }
    // --- stage B tile (32 x 16*NCT), transposed into column-major LDS -----
    for (int i = tid; i < TILE_K * 16 * NCT; i += 256) {
      int r = i / (16 * NCT);                  // K within tile
      int c = i % (16 * NCT);                  // output column
      float v = (c < Hout) ? W[(size_t)(k0 + r) * Hout + c] : 0.0f;
      Bs[c * LDT + r] = (__bf16)v;
    }
    __syncthreads();

    // --- A fragment: two contiguous b128 loads ----------------------------
    FragU fa;
    const int aoff = (m0 + M) * LDT + hs * 8;
    fa.u[0] = *(const v4u*)&As[aoff];          // K chunk [8hs, 8hs+8)
    fa.u[1] = *(const v4u*)&As[aoff + 16];     // K chunk [16+8hs, +8)

    #pragma unroll
    for (int n = 0; n < NCT; ++n) {
      FragU fb;
      const int boff = (n * 16 + M) * LDT + hs * 8;
      fb.u[0] = *(const v4u*)&Bs[boff];
      fb.u[1] = *(const v4u*)&Bs[boff + 16];
      acc[n] = __builtin_amdgcn_wmma_f32_16x16x32_bf16(
          false, fa.b, false, fb.b, (short)0, acc[n], false, false);
    }
    __syncthreads();
  }

  // --- store C: VGPR r <-> M = r + 8*hs, N = lane&15 -----------------------
  const int rowb = rowBase + m0 + 8 * hs;
  #pragma unroll
  for (int n = 0; n < NCT; ++n) {
    #pragma unroll
    for (int r = 0; r < 8; ++r) {
      int row = rowb + r;
      int col = n * 16 + M;
      if (row < Nrows && col < Hout)
        out[(size_t)row * Hout + col] = acc[n][r];
    }
  }
}

// ---------------------------------------------------------------------------
// Per-node attention logits: asrc[n] = h[n]·a_s, adst[n] = h[n]·a_d.
// One wave32 per node; also initializes segment max/sum buffers.
// ---------------------------------------------------------------------------
__global__ void k_alpha(const float* __restrict__ h, const float* __restrict__ a_s,
                        const float* __restrict__ a_d, float* __restrict__ asrc,
                        float* __restrict__ adst, float* __restrict__ m,
                        float* __restrict__ z, int n_nodes, int hdim)
{
  int gw   = (int)((blockIdx.x * blockDim.x + threadIdx.x) >> 5);
  int lane = threadIdx.x & 31;
  if (gw >= n_nodes) return;
  float s0 = 0.f, s1 = 0.f;
  for (int j = lane; j < hdim; j += 32) {
    float v = h[(size_t)gw * hdim + j];
    s0 += v * a_s[j];
    s1 += v * a_d[j];
  }
  #pragma unroll
  for (int off = 16; off > 0; off >>= 1) {
    s0 += __shfl_down(s0, off, 32);
    s1 += __shfl_down(s1, off, 32);
  }
  if (lane == 0) {
    asrc[gw] = s0;
    adst[gw] = s1;
    m[gw]    = -FLT_MAX;
    z[gw]    = 0.f;
  }
}

__device__ __forceinline__ void edge_sd(int e, const int* src, const int* dst,
                                        int E_, int& s, int& d)
{
  if (e < E_) { s = src[e]; d = dst[e]; }
  else        { s = d = e - E_; }          // self loops
}

// segment max of leaky(asrc[s]+adst[d], 0.2) over dst
__global__ void k_edge_max(const int* __restrict__ src, const int* __restrict__ dst,
                           const float* __restrict__ asrc, const float* __restrict__ adst,
                           float* __restrict__ m, int E_, int N_)
{
  int e = blockIdx.x * blockDim.x + threadIdx.x;
  if (e >= E_ + N_) return;
  int s, d; edge_sd(e, src, dst, E_, s, d);
  float x = asrc[s] + adst[d];
  x = (x >= 0.f) ? x : 0.2f * x;
  atomicMax(&m[d], x);                     // GLOBAL_ATOMIC_MAX_NUM_F32
}

// ex[e] = exp(e - m[dst]); z[dst] += ex
__global__ void k_edge_expsum(const int* __restrict__ src, const int* __restrict__ dst,
                              const float* __restrict__ asrc, const float* __restrict__ adst,
                              const float* __restrict__ m, float* __restrict__ z,
                              float* __restrict__ ex, int E_, int N_)
{
  int e = blockIdx.x * blockDim.x + threadIdx.x;
  if (e >= E_ + N_) return;
  int s, d; edge_sd(e, src, dst, E_, s, d);
  float x = asrc[s] + adst[d];
  x = (x >= 0.f) ? x : 0.2f * x;
  float ev = expf(x - m[d]);
  ex[e] = ev;
  atomicAdd(&z[d], ev);
}

// out[n,j] = bias[j] (+ resid[n,j])
__global__ void k_out_init(float* __restrict__ out, const float* __restrict__ bias,
                           const float* __restrict__ resid, int n_nodes, int hdim)
{
  int i = blockIdx.x * blockDim.x + threadIdx.x;
  if (i >= n_nodes * hdim) return;
  int col = i % hdim;
  float v = bias[col];
  if (resid) v += resid[i];
  out[i] = v;
}

// out[dst,:] += (ex[e]/z[dst]) * hp[src,:]  (one wave per edge)
__global__ void k_aggregate(const int* __restrict__ src, const int* __restrict__ dst,
                            const float* __restrict__ hp, const float* __restrict__ ex,
                            const float* __restrict__ z, float* __restrict__ out,
                            int E_, int N_, int hdim)
{
  int e    = (int)((blockIdx.x * blockDim.x + threadIdx.x) >> 5);
  int lane = threadIdx.x & 31;
  if (e >= E_ + N_) return;
  int s, d; edge_sd(e, src, dst, E_, s, d);
  float coef = ex[e] / z[d];
  for (int j = lane; j < hdim; j += 32)
    atomicAdd(&out[(size_t)d * hdim + j], coef * hp[(size_t)s * hdim + j]);
}

// ---------------------------------------------------------------------------
// BatchNorm (training stats, biased variance) + LeakyReLU(0.1)
// ---------------------------------------------------------------------------
__global__ void k_bn_zero(float* bsum, float* bsq)
{
  int i = threadIdx.x;
  if (i < 128) { bsum[i] = 0.f; bsq[i] = 0.f; }
}

__global__ void k_bn_stats(const float* __restrict__ h, float* __restrict__ bsum,
                           float* __restrict__ bsq, int n_nodes)
{
  int col = threadIdx.x;                    // 128 threads = columns
  int r0  = blockIdx.x * 256;
  int r1  = r0 + 256; if (r1 > n_nodes) r1 = n_nodes;
  float s = 0.f, q = 0.f;
  for (int r = r0; r < r1; ++r) {
    float v = h[(size_t)r * 128 + col];
    s += v; q += v * v;
  }
  atomicAdd(&bsum[col], s);
  atomicAdd(&bsq[col],  q);
}

__global__ void k_bn_apply(const float* __restrict__ h, const float* __restrict__ bsum,
                           const float* __restrict__ bsq, const float* __restrict__ g,
                           const float* __restrict__ be, float* __restrict__ out,
                           int n_nodes)
{
  int i = blockIdx.x * blockDim.x + threadIdx.x;
  if (i >= n_nodes * 128) return;
  int col = i & 127;
  float inv_n = 1.0f / (float)n_nodes;
  float mu  = bsum[col] * inv_n;
  float var = bsq[col] * inv_n - mu * mu;
  float v   = (h[i] - mu) * rsqrtf(var + 1e-5f) * g[col] + be[col];
  out[i] = (v >= 0.f) ? v : 0.1f * v;
}

__global__ void k_leaky_inplace(float* __restrict__ x, int n, float slope)
{
  int i = blockIdx.x * blockDim.x + threadIdx.x;
  if (i >= n) return;
  float v = x[i];
  x[i] = (v >= 0.f) ? v : slope * v;
}

// ---------------------------------------------------------------------------
// Host-side layer driver (launches only; graph-capture safe)
// ---------------------------------------------------------------------------
static void gat_layer(const float* in, int K, const float* W, const float* a_s,
                      const float* a_d, const float* bias, const float* resid,
                      float* hp, float* out, int Hout, int Nn, int Ee,
                      const int* src, const int* dst, float* asrc, float* adst,
                      float* m, float* z, float* ex, hipStream_t stream)
{
  const int Etot = Ee + Nn;
  const dim3 gg((Nn + TILE_M - 1) / TILE_M);
  if (Hout > 16)
    k_gemm_wmma<8><<<gg, 256, 0, stream>>>(in, W, hp, Nn, K, Hout);
  else
    k_gemm_wmma<1><<<gg, 256, 0, stream>>>(in, W, hp, Nn, K, Hout);
  k_alpha<<<dim3((Nn + 7) / 8), 256, 0, stream>>>(hp, a_s, a_d, asrc, adst, m, z, Nn, Hout);
  int eb = (Etot + 255) / 256;
  k_edge_max<<<dim3(eb), 256, 0, stream>>>(src, dst, asrc, adst, m, Ee, Nn);
  k_edge_expsum<<<dim3(eb), 256, 0, stream>>>(src, dst, asrc, adst, m, z, ex, Ee, Nn);
  k_out_init<<<dim3((Nn * Hout + 255) / 256), 256, 0, stream>>>(out, bias, resid, Nn, Hout);
  k_aggregate<<<dim3((Etot + 7) / 8), 256, 0, stream>>>(src, dst, hp, ex, z, out, Ee, Nn, Hout);
}

static void bn_leaky(const float* h, const float* g, const float* be, float* out,
                     float* bsum, float* bsq, int Nn, hipStream_t stream)
{
  k_bn_zero<<<dim3(1), 128, 0, stream>>>(bsum, bsq);
  k_bn_stats<<<dim3((Nn + 255) / 256), 128, 0, stream>>>(h, bsum, bsq, Nn);
  k_bn_apply<<<dim3((Nn * 128 + 255) / 256), 256, 0, stream>>>(h, bsum, bsq, g, be, out, Nn);
}

extern "C" void kernel_launch(void* const* d_in, const int* in_sizes, int n_in,
                              void* d_out, int out_size, void* d_ws, size_t ws_size,
                              hipStream_t stream)
{
  const int F_IN = 768, H = 128, C = 9, M_LAYERS = 3;
  const int Nn = in_sizes[0] / F_IN;   // 50000
  const int Ee = in_sizes[1] / 2;      // 800000

  const float* x    = (const float*)d_in[0];
  const int*   eidx = (const int*)  d_in[1];
  const float* W0   = (const float*)d_in[4];
  const float* a0s  = (const float*)d_in[5];
  const float* a0d  = (const float*)d_in[6];
  const float* b0   = (const float*)d_in[7];
  const float* Wm   = (const float*)d_in[8];
  const float* ams  = (const float*)d_in[9];
  const float* amd_ = (const float*)d_in[10];
  const float* bm   = (const float*)d_in[11];
  const float* W4   = (const float*)d_in[12];
  const float* a4s  = (const float*)d_in[13];
  const float* a4d  = (const float*)d_in[14];
  const float* b4   = (const float*)d_in[15];
  const float* gamma= (const float*)d_in[16];
  const float* beta = (const float*)d_in[17];

  const int* src = eidx;        // edge_index[0]
  const int* dst = eidx + Ee;   // edge_index[1]

  // workspace carve-up (256B aligned)
  size_t off = 0;
  char* base = (char*)d_ws;
  auto carve = [&](size_t nfloats) {
    float* p = (float*)(base + off);
    off += ((nfloats * sizeof(float) + 255) / 256) * 256;
    return p;
  };
  float* h    = carve((size_t)Nn * H);
  float* t    = carve((size_t)Nn * H);
  float* hp   = carve((size_t)Nn * H);
  float* asrc = carve(Nn);
  float* adst = carve(Nn);
  float* m    = carve(Nn);
  float* z    = carve(Nn);
  float* ex   = carve((size_t)(Ee + Nn));
  float* bsum = carve(H);
  float* bsq  = carve(H);
  (void)ws_size; (void)n_in; (void)out_size;

  // Layer 0: h = GAT(x; W0)
  gat_layer(x, F_IN, W0, a0s, a0d, b0, nullptr, hp, h, H, Nn, Ee,
            src, dst, asrc, adst, m, z, ex, stream);

  // Mid layers: t = leaky(BN(h)); h = GAT(t) + t
  for (int i = 0; i < M_LAYERS; ++i) {
    bn_leaky(h, gamma + i * H, beta + i * H, t, bsum, bsq, Nn, stream);
    gat_layer(t, H, Wm + (size_t)i * H * H, ams + i * H, amd_ + i * H,
              bm + i * H, t /*residual*/, hp, h, H, Nn, Ee,
              src, dst, asrc, adst, m, z, ex, stream);
  }

  // Final: t = leaky(BN(h)); out = leaky(GAT(t; W4), 0.1)
  bn_leaky(h, gamma + M_LAYERS * H, beta + M_LAYERS * H, t, bsum, bsq, Nn, stream);
  float* dout = (float*)d_out;
  gat_layer(t, H, W4, a4s, a4d, b4, nullptr, hp, dout, C, Nn, Ee,
            src, dst, asrc, adst, m, z, ex, stream);
  k_leaky_inplace<<<dim3((Nn * C + 255) / 256), 256, 0, stream>>>(dout, Nn * C, 0.1f);
}
